// PTBModel_47991964566287
// MI455X (gfx1250) — matrix operand order repaired and checked by reading, not compiled, not run on hardware
//
#include <hip/hip_runtime.h>
#include <hip/hip_bf16.h>

// ---- problem constants (PTB LSTM) ----
#define VOCAB  32000
#define EMBED  1024
#define HIDDEN 1024
#define NBATCH 32
#define SEQLEN 128
#define BT     (NBATCH * SEQLEN)   // 4096 rows
#define G4     (4 * HIDDEN)        // 4096 gate cols

typedef __attribute__((ext_vector_type(4)))  _Float16     v4h;
typedef __attribute__((ext_vector_type(8)))  _Float16     v8h;
typedef __attribute__((ext_vector_type(16))) _Float16     v16h;
typedef __attribute__((ext_vector_type(8)))  float        v8f;
typedef __attribute__((ext_vector_type(4)))  unsigned int v4u;
typedef __attribute__((ext_vector_type(4)))  int          v4i;
typedef __attribute__((ext_vector_type(8)))  int          v8i;

#ifndef __has_builtin
#define __has_builtin(x) 0
#endif
#if __has_builtin(__builtin_amdgcn_tensor_load_to_lds) && \
    __has_builtin(__builtin_amdgcn_s_wait_tensorcnt)
#define HAVE_TDM 1
#else
#define HAVE_TDM 0
#endif
// ROCm 7.2 clang-22 ships the 5-arg tensor builtin; the therock toolchain
// (which is the only one shipping the TDM header) ships the 6-arg form.
#if __has_include(<hip/amd_detail/amd_gfx1250_TDM.h>)
#define TDM_ARG6 1
#else
#define TDM_ARG6 0
#endif

__device__ __forceinline__ float sigmoidf_(float x) {
  return 1.0f / (1.0f + __expf(-x));
}

#if HAVE_TDM
// ---------------------------------------------------------------------------
// Issue one TDM 2D tile load (Global -> LDS), per CDNA5 ISA
// D# layout.  tile_w/row_stride/tensor dims in 2-byte (f16) elements.
// Must be executed by exactly one wave per workgroup (EXEC ignored by TDM).
// ---------------------------------------------------------------------------
__device__ __forceinline__ void tdm_load_tile(const _Float16* gsrc, void* lds_dst,
                                              unsigned tile_w, unsigned tile_h,
                                              unsigned row_stride,
                                              unsigned tens_w, unsigned tens_h) {
  // LDS aperture: generic pointer low 32 bits are the LDS byte offset.
  const unsigned lds_addr = (unsigned)(size_t)lds_dst;
  const unsigned long long ga = (unsigned long long)(size_t)gsrc;
  v4u g0;
  g0[0] = 1u;                                             // count=1, load D#
  g0[1] = lds_addr;                                       // lds_addr
  g0[2] = (unsigned)(ga & 0xffffffffu);                   // global_addr[31:0]
  g0[3] = (unsigned)((ga >> 32) & 0x1ffffffu) | (2u << 30); // [56:32] | type=2
  const unsigned long long s0 = (unsigned long long)row_stride;
  v8i g1;
  g1[0] = (int)(1u << 16);                                // data_size=1 (2 bytes)
  g1[1] = (int)((tens_w & 0xffffu) << 16);                // tensor_dim0[15:0]
  g1[2] = (int)(((tens_h & 0xffffu) << 16) | (tens_w >> 16)); // td1[15:0]|td0[31:16]
  g1[3] = (int)(((tile_w & 0xffffu) << 16) | (tens_h >> 16)); // tile_dim0|td1[31:16]
  g1[4] = (int)(tile_h & 0xffffu);                        // tile_dim1 (tile_dim2=0)
  g1[5] = (int)(unsigned)(s0 & 0xffffffffu);              // dim0_stride[31:0]
  g1[6] = (int)(unsigned)((s0 >> 32) & 0xffffu);          // dim0_stride[47:32]
  g1[7] = 0;                                              // dim1_stride (unused, 2D)
  v4i z4 = {0, 0, 0, 0};
#if TDM_ARG6
  v8i z8 = {0, 0, 0, 0, 0, 0, 0, 0};
  __builtin_amdgcn_tensor_load_to_lds(g0, g1, z4, z4, z8, 0);
#else
  __builtin_amdgcn_tensor_load_to_lds(g0, g1, z4, z4, 0);
#endif
}
#endif  // HAVE_TDM

// ---------------------------------------------------------------------------
// Transpose + cast: in f32 [R x C] row-major  ->  out f16 [C x R] row-major.
// ---------------------------------------------------------------------------
__global__ __launch_bounds__(256)
void k_transpose_cast(const float* __restrict__ in, _Float16* __restrict__ out,
                      int R, int C) {
  __shared__ float tile[32][33];
  const int k0 = blockIdx.y * 32, n0 = blockIdx.x * 32;
  const int tx = threadIdx.x, ty = threadIdx.y;   // block (32, 8)
#pragma unroll
  for (int i = 0; i < 32; i += 8)
    tile[ty + i][tx] = in[(size_t)(k0 + ty + i) * C + (n0 + tx)];
  __syncthreads();
#pragma unroll
  for (int i = 0; i < 32; i += 8)
    out[(size_t)(n0 + ty + i) * R + (k0 + tx)] = (_Float16)tile[tx][ty + i];
}

// ---------------------------------------------------------------------------
// Embedding gather + cast to f16: X[bt][:] = emb[seq[bt]][:]
// ---------------------------------------------------------------------------
__global__ __launch_bounds__(256)
void k_embed(const int* __restrict__ seq, const float* __restrict__ emb,
             _Float16* __restrict__ X) {
  const int bt = blockIdx.x;
  const int row = seq[bt];
  const float4* src = (const float4*)(emb + (size_t)row * EMBED);
  float4 v = src[threadIdx.x];                      // 256 thr * 4 = 1024
  v4h o = {(_Float16)v.x, (_Float16)v.y, (_Float16)v.z, (_Float16)v.w};
  *(v4h*)&X[(size_t)bt * EMBED + threadIdx.x * 4] = o;
}

// ---------------------------------------------------------------------------
// Tiled WMMA GEMM:  C[M,N] = A[M,K](f16,row) * Bt[N,K](f16,row) + bias[N]
// Block 256 thr = 8 waves, block tile 64(M) x 128(N), K step 32.
// Both A (64x32) and B (128x32) tiles are double-buffered in LDS and fetched
// by the Tensor Data Mover: each iteration wave 0 issues the (A,B) descriptor
// pair for chunk kk+32, then s_wait_tensorcnt(2) guarantees the current pair
// has landed before the barrier releases the consumers.  This removes the 2x
// duplicate B fetches across the two M-waves and overlaps DMA with WMMA.
// ---------------------------------------------------------------------------
template <typename OutT>
__global__ __launch_bounds__(256)
void k_gemm(const _Float16* __restrict__ A, const _Float16* __restrict__ Bt,
            const float* __restrict__ bias, OutT* __restrict__ C,
            int M, int N, int K) {
  __shared__ __align__(32) _Float16 As[2][64 * 32];   // 2 x 4 KB
#if HAVE_TDM
  __shared__ __align__(32) _Float16 Bs[2][128 * 32];  // 2 x 8 KB
#endif
  const int tid  = threadIdx.x;
  const int lane = tid & 31;
  const int w    = tid >> 5;
  const int wm   = w & 1;
  const int wn   = w >> 1;
  const int l15  = lane & 15;
  const int lh   = lane >> 4;          // lane half (0/1)
  const int akb  = lh * 16;            // A k-base within 32-chunk
  const int bkb  = lh * 16;            // B k-base within 32-chunk
  const size_t m_blk = (size_t)blockIdx.y * 64;
  const size_t n_blk = (size_t)blockIdx.x * 128;
  const int arow = tid >> 2;           // 0..63 (fallback A stage)
  const int acol = (tid & 3) * 8;

  v8f acc[2][2] = {};

#if HAVE_TDM
  if (w == 0) {
    tdm_load_tile(&A[m_blk * K], &As[0][0], 32, 64, (unsigned)K,
                  (unsigned)K, (unsigned)M);
    tdm_load_tile(&Bt[n_blk * K], &Bs[0][0], 32, 128, (unsigned)K,
                  (unsigned)K, (unsigned)N);
  }
#endif

  int cur = 0;
  for (int kk = 0; kk < K; kk += 32, cur ^= 1) {
#if HAVE_TDM
    if (w == 0) {
      if (kk + 32 < K) {
        tdm_load_tile(&A[m_blk * K + kk + 32], &As[cur ^ 1][0], 32, 64,
                      (unsigned)K, (unsigned)K, (unsigned)M);
        tdm_load_tile(&Bt[n_blk * K + kk + 32], &Bs[cur ^ 1][0], 32, 128,
                      (unsigned)K, (unsigned)K, (unsigned)N);
        __builtin_amdgcn_s_wait_tensorcnt(2);   // current (A,B) pair landed
      } else {
        __builtin_amdgcn_s_wait_tensorcnt(0);   // last pair landed
      }
    }
    __syncthreads();                            // publish tiles to all waves
#else
    __syncthreads();
    *(v8h*)&As[cur][arow * 32 + acol] =
        *(const v8h*)&A[(m_blk + arow) * K + kk + acol];
    __syncthreads();
#endif

    v16h a[2], b[2];
#pragma unroll
    for (int sm = 0; sm < 2; ++sm) {
      const int m = wm * 32 + sm * 16 + l15;
      v8h lo = *(const v8h*)&As[cur][m * 32 + akb];
      v8h hi = *(const v8h*)&As[cur][m * 32 + akb + 8];
      a[sm] = __builtin_shufflevector(lo, hi, 0, 1, 2, 3, 4, 5, 6, 7,
                                      8, 9, 10, 11, 12, 13, 14, 15);
    }
#pragma unroll
    for (int sn = 0; sn < 2; ++sn) {
#if HAVE_TDM
      const int nl = wn * 32 + sn * 16 + l15;     // row within B LDS tile
      b[sn] = *(const v16h*)&Bs[cur][nl * 32 + bkb];
#else
      const size_t n = n_blk + wn * 32 + sn * 16 + l15;
      b[sn] = *(const v16h*)&Bt[n * K + kk + bkb];
      __builtin_prefetch(&Bt[n * K + kk + 32 + bkb], 0, 1);
#endif
    }
#pragma unroll
    for (int sm = 0; sm < 2; ++sm)
#pragma unroll
      for (int sn = 0; sn < 2; ++sn)
        acc[sm][sn] = __builtin_amdgcn_wmma_f32_16x16x32_f16(
            false, a[sm], false, b[sn], (short)0, acc[sm][sn], false, false);

#if HAVE_TDM
    __syncthreads();   // all reads of tiles done before next TDM overwrite
#endif
  }

  // Epilogue: lane holds rows mbase..mbase+7 of its 16x16 tile, col = l15.
#pragma unroll
  for (int sm = 0; sm < 2; ++sm) {
#pragma unroll
    for (int sn = 0; sn < 2; ++sn) {
      const size_t n = n_blk + wn * 32 + sn * 16 + l15;
      const float bv = bias ? bias[n] : 0.0f;
      const size_t m0 = m_blk + wm * 32 + sm * 16 + lh * 8;
#pragma unroll
      for (int r = 0; r < 8; ++r)
        C[(m0 + r) * (size_t)N + n] = (OutT)(acc[sm][sn][r] + bv);
    }
  }
}

// ---------------------------------------------------------------------------
// Fused LSTM recurrent step (one t, one layer):
//   gates = gates_x[.,t,:] + h_{t-1} @ Wh ; cell update ; writes h_t, c.
// Grid: 64 blocks (16 hidden cols each) x 64 thr (2 waves: batch 0-15/16-31).
// All four gate B-fragments are loaded before the four WMMAs so the loads
// clause together; prefetches are unconditional (branch-free inner loop --
// the 64 B overrun on the last iteration stays inside the workspace and
// speculative DEV-scope prefetch drops bad translations silently).
// ---------------------------------------------------------------------------
__global__ __launch_bounds__(64)
void k_lstm_step(_Float16* __restrict__ hlay,        // [BT, HIDDEN] f16 (in/out)
                 const _Float16* __restrict__ Wht,   // [G4, HIDDEN] f16
                 const _Float16* __restrict__ gx,    // [BT, G4] f16 (x-proj + bias)
                 float* __restrict__ c,              // [NBATCH, HIDDEN] f32
                 int t) {
  const int tid  = threadIdx.x;
  const int lane = tid & 31;
  const int wv   = tid >> 5;          // 0/1 -> batch rows 0-15 / 16-31
  const int hc   = blockIdx.x * 16;   // hidden column base
  const int l15  = lane & 15;
  const int lh   = lane >> 4;

  v8f acc[4] = {};
  if (t > 0) {
    for (int kk = 0; kk < HIDDEN; kk += 32) {
      const int m = wv * 16 + l15;                    // batch row
      const _Float16* ap =
          hlay + ((size_t)m * SEQLEN + (t - 1)) * HIDDEN + kk;
      v8h lo = *(const v8h*)(ap + lh * 16);
      v8h hi = *(const v8h*)(ap + lh * 16 + 8);
      v16h a = __builtin_shufflevector(lo, hi, 0, 1, 2, 3, 4, 5, 6, 7,
                                       8, 9, 10, 11, 12, 13, 14, 15);
      v16h b[4];
#pragma unroll
      for (int g = 0; g < 4; ++g) {
        const size_t n = (size_t)g * HIDDEN + hc + l15;
        b[g] = *(const v16h*)&Wht[n * HIDDEN + kk + lh * 16];
        __builtin_prefetch(&Wht[n * HIDDEN + kk + 32 + lh * 16], 0, 1);
      }
#pragma unroll
      for (int g = 0; g < 4; ++g)
        acc[g] = __builtin_amdgcn_wmma_f32_16x16x32_f16(
            false, a, false, b[g], (short)0, acc[g], false, false);
    }
  }

  const int n = hc + l15;
#pragma unroll
  for (int r = 0; r < 8; ++r) {
    const int m = wv * 16 + lh * 8 + r;               // batch index
    const size_t gxrow = ((size_t)m * SEQLEN + t) * (size_t)G4;
    const float gi = acc[0][r] + (float)gx[gxrow + 0 * HIDDEN + n];
    const float gj = acc[1][r] + (float)gx[gxrow + 1 * HIDDEN + n];
    const float gf = acc[2][r] + (float)gx[gxrow + 2 * HIDDEN + n];
    const float go = acc[3][r] + (float)gx[gxrow + 3 * HIDDEN + n];
    const float cold = (t > 0) ? c[(size_t)m * HIDDEN + n] : 0.0f;
    const float cn = sigmoidf_(gf + 1.0f) * cold + sigmoidf_(gi) * tanhf(gj);
    const float hn = sigmoidf_(go) * tanhf(cn);
    c[(size_t)m * HIDDEN + n] = cn;
    hlay[((size_t)m * SEQLEN + t) * HIDDEN + n] = (_Float16)hn;
  }
}

// ---------------------------------------------------------------------------
extern "C" void kernel_launch(void* const* d_in, const int* in_sizes, int n_in,
                              void* d_out, int out_size, void* d_ws, size_t ws_size,
                              hipStream_t stream) {
  (void)in_sizes; (void)n_in; (void)out_size; (void)ws_size;
  const int*   seq = (const int*)d_in[0];
  const float* emb = (const float*)d_in[1];
  const float* W0  = (const float*)d_in[2];
  const float* b0  = (const float*)d_in[3];
  const float* W1  = (const float*)d_in[4];
  const float* b1  = (const float*)d_in[5];
  const float* Wd  = (const float*)d_in[6];
  const float* bd  = (const float*)d_in[7];
  float* out = (float*)d_out;

  // scratch carving (256B aligned)
  char* p = (char*)d_ws;
  auto carve = [&](size_t bytes) -> char* {
    char* r = p; p += (bytes + 255) & ~(size_t)255; return r;
  };
  _Float16* X    = (_Float16*)carve((size_t)BT * EMBED * 2);
  _Float16* Wx0t = (_Float16*)carve((size_t)G4 * EMBED * 2);
  _Float16* Wh0t = (_Float16*)carve((size_t)G4 * HIDDEN * 2);
  _Float16* Wx1t = (_Float16*)carve((size_t)G4 * HIDDEN * 2);
  _Float16* Wh1t = (_Float16*)carve((size_t)G4 * HIDDEN * 2);
  _Float16* Wdt  = (_Float16*)carve((size_t)VOCAB * HIDDEN * 2);
  _Float16* GX   = (_Float16*)carve((size_t)BT * G4 * 2);   // reused per layer
  _Float16* H0   = (_Float16*)carve((size_t)BT * HIDDEN * 2);
  _Float16* H1   = (_Float16*)carve((size_t)BT * HIDDEN * 2);
  float*    C0   = (float*)carve((size_t)NBATCH * HIDDEN * 4);
  float*    C1   = (float*)carve((size_t)NBATCH * HIDDEN * 4);

  const dim3 tb(32, 8);
  // weight transposes (f32 -> f16, [K,N] -> [N,K])
  k_transpose_cast<<<dim3(G4 / 32, EMBED / 32), tb, 0, stream>>>(W0, Wx0t, EMBED, G4);
  k_transpose_cast<<<dim3(G4 / 32, HIDDEN / 32), tb, 0, stream>>>(W0 + (size_t)EMBED * G4, Wh0t, HIDDEN, G4);
  k_transpose_cast<<<dim3(G4 / 32, HIDDEN / 32), tb, 0, stream>>>(W1, Wx1t, HIDDEN, G4);
  k_transpose_cast<<<dim3(G4 / 32, HIDDEN / 32), tb, 0, stream>>>(W1 + (size_t)HIDDEN * G4, Wh1t, HIDDEN, G4);
  k_transpose_cast<<<dim3(VOCAB / 32, HIDDEN / 32), tb, 0, stream>>>(Wd, Wdt, HIDDEN, VOCAB);

  // embedding gather
  k_embed<<<BT, 256, 0, stream>>>(seq, emb, X);

  // layer 0: input projection (all timesteps), then 128 fused recurrent steps
  k_gemm<_Float16><<<dim3(G4 / 128, BT / 64), 256, 0, stream>>>(X, Wx0t, b0, GX, BT, G4, EMBED);
  for (int t = 0; t < SEQLEN; ++t)
    k_lstm_step<<<HIDDEN / 16, 64, 0, stream>>>(H0, Wh0t, GX, C0, t);

  // layer 1
  k_gemm<_Float16><<<dim3(G4 / 128, BT / 64), 256, 0, stream>>>(H0, Wx1t, b1, GX, BT, G4, HIDDEN);
  for (int t = 0; t < SEQLEN; ++t)
    k_lstm_step<<<HIDDEN / 16, 64, 0, stream>>>(H1, Wh1t, GX, C1, t);

  // final dense projection -> f32 logits [BT, VOCAB]
  k_gemm<float><<<dim3(VOCAB / 128, BT / 64), 256, 0, stream>>>(H1, Wdt, bd, out, BT, VOCAB, HIDDEN);
}
